// GRU_42820823941719
// MI455X (gfx1250) — compile-verified, matrix operand order checked by reading, not run on hardware
//
#include <hip/hip_runtime.h>
#include <hip/hip_bf16.h>

// ---------------- problem constants (from reference) ----------------
constexpr int B  = 512;
constexpr int S  = 128;
constexpr int P  = 32;
constexpr int F  = 64;
constexpr int O  = 64;
constexpr int H  = 1024;
constexpr int IN = 128;      // F + O
constexpr int G3 = 3 * H;    // 3072

// ---------------- types ----------------
typedef __bf16 bf16;
typedef __attribute__((ext_vector_type(16))) __bf16 v16bf;
typedef __attribute__((ext_vector_type(8)))  float  v8f;
typedef __attribute__((ext_vector_type(4)))  unsigned int v4u;

__device__ __forceinline__ bf16 f2bf(float f) { return (bf16)f; }

// WMMA lane-packed position of logical K index u inside its 32-wide chunk:
// store order per chunk = {0-7,16-23 | 8-15,24-31} so each lane half reads
// its 16 operand values as one contiguous 32-byte vector.
__host__ __device__ __forceinline__ int pk(int u) {
    return (u & ~31) + (((u >> 3) & 1) << 4) + (u & 7) + (((u >> 4) & 1) << 3);
}

__device__ __forceinline__ v8f wmma_bf16(v16bf a, v16bf b, v8f c) {
    return __builtin_amdgcn_wmma_f32_16x16x32_bf16(
        false, a, false, b, (short)0, c, false, false);
}

__device__ __forceinline__ float sigmoidf_(float x) {
    return 1.0f / (1.0f + __expf(-x));
}

// ---------------- prep: one-shot bf16 conversion + packing + h0 ----------------
__global__ void prep_kernel(const float* __restrict__ feats, const float* __restrict__ labels,
                            const float* __restrict__ Wi, const float* __restrict__ Wh,
                            const float* __restrict__ Wd,
                            bf16* __restrict__ Wip, bf16* __restrict__ Whp,
                            bf16* __restrict__ Wdp,
                            float* __restrict__ h32, bf16* __restrict__ hbf,
                            bf16* __restrict__ Xenc, bf16* __restrict__ Xdec) {
    const long NI  = (long)G3 * IN;
    const long NH  = (long)G3 * H;
    const long ND  = (long)O * H;
    const long NB  = (long)B * H;
    const long NXE = (long)S * B * IN;
    const long NXD = (long)(P - 1) * B * F;
    const long total = NI + NH + ND + NB + NXE + NXD;
    const long sFeat = (long)(S + P - 1) * F;
    const long sLab  = (long)S * O;
    const long stride = (long)gridDim.x * blockDim.x;
    for (long i = (long)blockIdx.x * blockDim.x + threadIdx.x; i < total; i += stride) {
        if (i < NI) {
            int j = (int)(i / IN), k = (int)(i % IN);
            Wip[(long)j * IN + pk(k)] = f2bf(Wi[i]);
        } else if (i < NI + NH) {
            long q = i - NI; int j = (int)(q / H), k = (int)(q % H);
            Whp[(long)j * H + pk(k)] = f2bf(Wh[q]);
        } else if (i < NI + NH + ND) {
            long q = i - NI - NH; int j = (int)(q / H), k = (int)(q % H);
            Wdp[(long)j * H + pk(k)] = f2bf(Wd[q]);
        } else if (i < NI + NH + ND + NB) {
            long q = i - NI - NH - ND;
            h32[q] = 0.0f; hbf[q] = f2bf(0.0f);
        } else if (i < NI + NH + ND + NB + NXE) {
            long q = i - NI - NH - ND - NB;
            int t = (int)(q / ((long)B * IN));
            int r = (int)(q % ((long)B * IN));
            int b = r / IN, u = r % IN;
            float v = (u < F) ? feats[(long)b * sFeat + (long)t * F + u]
                              : labels[(long)b * sLab + (long)t * O + (u - F)];
            Xenc[(long)t * B * IN + (long)b * IN + pk(u)] = f2bf(v);
        } else {
            long q = i - NI - NH - ND - NB - NXE;
            int t = (int)(q / ((long)B * F));
            int r = (int)(q % ((long)B * F));
            int b = r / F, c = r % F;
            Xdec[(long)t * B * IN + (long)b * IN + pk(c)] =
                f2bf(feats[(long)b * sFeat + (long)(S + t) * F + c]);
        }
    }
}

// ---------------- one GRU timestep ----------------
// Block = 4 waves (128 threads). Each wave owns a 16-row batch subtile
// (M=64 per block); all waves share u-tile of 32 hidden units (x 3 gates).
// Weight chunks (96 rows x 32 K, WMMA-packed) are staged into LDS once per
// block and read by all 4 waves -> 4x less L2 weight traffic. Double-buffered.
#define NT 2
#define MW 4
__global__ __launch_bounds__(128)
void gru_step_kernel(const bf16*  __restrict__ xbf,   // [B][IN] packed
                     const bf16*  __restrict__ hbf, const float* __restrict__ h32,
                     const bf16*  __restrict__ Wip, const bf16* __restrict__ Whp,
                     const float* __restrict__ bi,  const float* __restrict__ bh,
                     float* __restrict__ h32o, bf16* __restrict__ hbfo) {
    __shared__ v4u sB[2][96 * 4];                 // 2 x 6KB: 96 rows x 64B
    const int tid  = threadIdx.x;
    const int lane = tid & 31, wv = tid >> 5;
    const int hh   = lane >> 4, l16 = lane & 15;
    const int bb   = blockIdx.x * (16 * MW) + wv * 16;
    const int u0   = blockIdx.y * (16 * NT);
    const int arow = bb + l16;

    v8f accr[NT], accz[NT], accn[NT], acchn[NT];
#pragma unroll
    for (int nt = 0; nt < NT; ++nt) {
        accr[nt] = (v8f)0.0f; accz[nt] = (v8f)0.0f;
        accn[nt] = (v8f)0.0f; acchn[nt] = (v8f)0.0f;
    }

    const int NKX = IN / 32, NKH = H / 32, NK = NKX + NKH;

    // stage chunk c of the weight matrices into LDS buffer `buf`
    auto stage = [&](int buf, int c) {
        const bf16* Wb = (c < NKX) ? Wip : Whp;
        const long  rs = (c < NKX) ? IN : H;
        const int   kb = ((c < NKX) ? c : (c - NKX)) * 32;
#pragma unroll
        for (int s = 0; s < 3; ++s) {             // 384 16B-chunks / 128 thr
            const int id = tid + 128 * s;
            const int r = id >> 2, p = id & 3;    // row 0..95, 16B part 0..3
            const int g = r >> 5, cc = r & 31;    // gate, column-in-tile
            const v4u* src = (const v4u*)(Wb + ((long)g * H + u0 + cc) * rs + kb) + p;
            sB[buf][(r << 2) + p] = *src;
        }
    };

    stage(0, 0);
    __syncthreads();
    int buf = 0;
    for (int c = 0; c < NK; ++c) {
        if (c + 1 < NK) stage(buf ^ 1, c + 1);
        // A operand: 32B contiguous per lane (pre-packed)
        v16bf a;
        if (c < NKX) a = *(const v16bf*)(xbf + (long)arow * IN + c * 32 + hh * 16);
        else         a = *(const v16bf*)(hbf + (long)arow * H + (c - NKX) * 32 + hh * 16);
#pragma unroll
        for (int nt = 0; nt < NT; ++nt) {
            const int rbase = nt * 16 + l16;
            v16bf b0 = *(const v16bf*)(&sB[buf][((0 * 32 + rbase) << 2) + (hh << 1)]);
            v16bf b1 = *(const v16bf*)(&sB[buf][((1 * 32 + rbase) << 2) + (hh << 1)]);
            v16bf b2 = *(const v16bf*)(&sB[buf][((2 * 32 + rbase) << 2) + (hh << 1)]);
            accr[nt] = wmma_bf16(a, b0, accr[nt]);
            accz[nt] = wmma_bf16(a, b1, accz[nt]);
            if (c < NKX) accn[nt]  = wmma_bf16(a, b2, accn[nt]);
            else         acchn[nt] = wmma_bf16(a, b2, acchn[nt]);
        }
        __syncthreads();                          // staging + compute both done
        buf ^= 1;
    }

    // ---- fused GRU gates; C layout: VGPR v -> M = v + 8*half, N = lane%16 ----
#pragma unroll
    for (int nt = 0; nt < NT; ++nt) {
        const int u = u0 + nt * 16 + l16;
        const int up = pk(u);                     // packed column for bf16 h-out
        const float br_  = bi[u]         + bh[u];
        const float bz_  = bi[H + u]     + bh[H + u];
        const float bin_ = bi[2 * H + u];
        const float bhn_ = bh[2 * H + u];
#pragma unroll
        for (int v = 0; v < 8; ++v) {
            const int brow = bb + hh * 8 + v;
            const float r  = sigmoidf_(accr[nt][v] + br_);
            const float z  = sigmoidf_(accz[nt][v] + bz_);
            const float n  = tanhf(accn[nt][v] + bin_ + r * (acchn[nt][v] + bhn_));
            const float hp = h32[(long)brow * H + u];
            const float hn = (1.0f - z) * n + z * hp;
            h32o[(long)brow * H + u]  = hn;
            hbfo[(long)brow * H + up] = f2bf(hn);
        }
    }
}

// ---------------- dense projection ps = h @ Wd^T + bd ----------------
__global__ __launch_bounds__(32)
void dense_step_kernel(const bf16* __restrict__ hbf, const bf16* __restrict__ Wdp,
                       const float* __restrict__ bd, float* __restrict__ ps,
                       bf16* __restrict__ xnext /* Xdec[t+1] base or nullptr */) {
    const int lane = threadIdx.x;
    const int hh   = lane >> 4;
    const int l16  = lane & 15;
    const int bb   = blockIdx.x * 16;
    const int o0   = blockIdx.y * 16;
    const int arow = bb + l16;

    v8f acc = (v8f)0.0f;
#pragma unroll 4
    for (int kt = 0; kt < H / 32; ++kt) {
        const int koff = kt * 32 + hh * 16;
        v16bf a = *(const v16bf*)(hbf + (long)arow * H + koff);
        v16bf b = *(const v16bf*)(Wdp + (long)(o0 + l16) * H + koff);
        acc = wmma_bf16(a, b, acc);
    }
    const int o  = o0 + l16;
    const float bo = bd[o];
    const int up = pk(F + o);
#pragma unroll
    for (int v = 0; v < 8; ++v) {
        const int brow = bb + hh * 8 + v;
        const float val = acc[v] + bo;
        ps[(long)brow * O + o] = val;
        if (xnext) xnext[(long)brow * IN + up] = f2bf(val);
    }
}

// ---------------- host-side orchestration ----------------
extern "C" void kernel_launch(void* const* d_in, const int* in_sizes, int n_in,
                              void* d_out, int out_size, void* d_ws, size_t ws_size,
                              hipStream_t stream) {
    (void)in_sizes; (void)n_in; (void)out_size; (void)ws_size;

    const float* feats  = (const float*)d_in[0]; // (B, S+P-1, F)
    const float* labels = (const float*)d_in[1]; // (B, S, O)
    const float* Wi     = (const float*)d_in[2]; // (3H, IN)
    const float* Wh     = (const float*)d_in[3]; // (3H, H)
    const float* bi     = (const float*)d_in[4];
    const float* bh     = (const float*)d_in[5];
    const float* Wd     = (const float*)d_in[6]; // (O, H)
    const float* bd     = (const float*)d_in[7];
    float* out = (float*)d_out;                  // (P, B, O)

    auto alignup = [](size_t x) { return (x + 255) & ~(size_t)255; };
    char* w = (char*)d_ws;
    bf16* Wip  = (bf16*)w; w += alignup((size_t)G3 * IN * sizeof(bf16));
    bf16* Whp  = (bf16*)w; w += alignup((size_t)G3 * H  * sizeof(bf16));
    bf16* Wdp  = (bf16*)w; w += alignup((size_t)O  * H  * sizeof(bf16));
    float* h32[2]; bf16* hbf[2];
    h32[0] = (float*)w; w += alignup((size_t)B * H * sizeof(float));
    h32[1] = (float*)w; w += alignup((size_t)B * H * sizeof(float));
    hbf[0] = (bf16*)w;  w += alignup((size_t)B * H * sizeof(bf16));
    hbf[1] = (bf16*)w;  w += alignup((size_t)B * H * sizeof(bf16));
    bf16* Xenc = (bf16*)w; w += alignup((size_t)S * B * IN * sizeof(bf16));
    bf16* Xdec = (bf16*)w; w += alignup((size_t)(P - 1) * B * IN * sizeof(bf16));

    // 1) one-shot conversion / packing / init
    prep_kernel<<<4096, 256, 0, stream>>>(feats, labels, Wi, Wh, Wd,
                                          Wip, Whp, Wdp, h32[0], hbf[0], Xenc, Xdec);

    const dim3 gGru(B / (16 * MW), H / (16 * NT));   // (8, 32)
    const dim3 gDen(B / 16, O / 16);

    int cur = 0;
    // 2) encoder: 128 sequential GRU steps
    for (int t = 0; t < S; ++t) {
        gru_step_kernel<<<gGru, 128, 0, stream>>>(
            Xenc + (long)t * B * IN,
            hbf[cur], h32[cur], Wip, Whp, bi, bh, h32[1 - cur], hbf[1 - cur]);
        cur ^= 1;
    }

    // 3) first prediction: out[0] and bf16 ps into Xdec[0]
    dense_step_kernel<<<gDen, 32, 0, stream>>>(hbf[cur], Wdp, bd, out, Xdec);

    // 4) decoder: 31 sequential steps
    for (int t = 0; t < P - 1; ++t) {
        gru_step_kernel<<<gGru, 128, 0, stream>>>(
            Xdec + (long)t * B * IN,
            hbf[cur], h32[cur], Wip, Whp, bi, bh, h32[1 - cur], hbf[1 - cur]);
        cur ^= 1;
        dense_step_kernel<<<gDen, 32, 0, stream>>>(
            hbf[cur], Wdp, bd, out + (long)(t + 1) * B * O,
            (t < P - 2) ? (Xdec + (long)(t + 1) * B * IN) : nullptr);
    }
}
